// Model_10651518894282
// MI455X (gfx1250) — matrix operand order repaired and checked by reading, not compiled, not run on hardware
//
#include <hip/hip_runtime.h>
#include <hip/hip_bf16.h>

#define DM   512
#define DI   1024
#define DS   16
#define DTR  32
#define NB   2
#define LSEQ 1000
#define NDIR 6
#define BL   (NB*LSEQ)   // 2000 rows per direction
#define CH   8           // scan chunks
#define LC   (LSEQ/CH)   // 125 steps per chunk

typedef __attribute__((ext_vector_type(16))) _Float16 v16h;
typedef __attribute__((ext_vector_type(8)))  _Float16 v8h;
typedef __attribute__((ext_vector_type(8)))  float    v8f;

#if defined(__AMDGCN__) && __has_builtin(__builtin_amdgcn_global_load_async_to_lds_b128) && \
    __has_builtin(__builtin_amdgcn_s_wait_asynccnt)
#define HAVE_ASYNC_LDS 1
// exact parameter types per hipcc diagnostic: int4 AS(1)* / int4 AS(3)*
typedef __attribute__((__vector_size__(4 * sizeof(int)))) int v4i;
typedef __attribute__((address_space(1))) v4i as1_v4i;
typedef __attribute__((address_space(3))) v4i as3_v4i;
#else
#define HAVE_ASYNC_LDS 0
#endif

// ---------------------------------------------------------------------------
// WMMA fragment loaders (contiguous k order shared by A and B, wave32)
// A 16x32 f16: lane m(0-15) row m K=0..15 ; lane 16+m row m K=16..31
__device__ __forceinline__ v16h load_a_frag(const _Float16* __restrict__ A,
                                            long lda, int row0, int k0, int lane) {
    const _Float16* r = A + (long)(row0 + (lane & 15)) * lda + k0 + ((lane >> 4) * 16);
    v8h lo = *(const v8h*)(r);
    v8h hi = *(const v8h*)(r + 8);
    v16h out;
#pragma unroll
    for (int i = 0; i < 8; ++i) { out[i] = lo[i]; out[8 + i] = hi[i]; }
    return out;
}
// B 32x16 f16 from weight W (N x K row-major): lane n: K=0..15 ; lane 16+n: K=16..31
__device__ __forceinline__ v16h load_b_frag(const _Float16* __restrict__ W,
                                            long ldw, int n0, int k0, int lane) {
    const _Float16* r = W + (long)(n0 + (lane & 15)) * ldw + k0 + ((lane >> 4) * 16);
    return *(const v16h*)r;
}

// ---------------------------------------------------------------------------
// C = A * W^T. Block = 4 waves; each wave owns a 16x64 tile (4 accumulators),
// all 4 waves share the same 64x32 B tile, staged in LDS via async-to-LDS
// (double buffered) when available. EPI: 0 = plain f32, 1 = +bias, softplus.
template <int EPI>
__global__ __launch_bounds__(128)
void gemm_wmma_f16(const _Float16* __restrict__ A, long strideA, long lda,
                   const _Float16* __restrict__ W, long strideW, long ldw,
                   float* __restrict__ C, long strideC, long ldc,
                   const float* __restrict__ bias, long strideBias,
                   int M, int N, int K) {
    __shared__ _Float16 bsm[2][64 * 32];   // two 4KB B tiles
    const int dir = blockIdx.z;
    A += (long)dir * strideA;
    W += (long)dir * strideW;
    C += (long)dir * strideC;
    const int tid  = threadIdx.x;          // 0..127
    const int wid  = tid >> 5;
    const int lane = tid & 31;
    const int row0   = (blockIdx.x * 4 + wid) * 16;
    const int n0base = blockIdx.y * 64;
    const bool active = (row0 < M);
    const int rowA = active ? row0 : (M - 16);   // clamp: keep all lanes busy/valid
    const int ksteps = K / 32;

    v8f acc[4] = {};

#if HAVE_ASYNC_LDS
    auto issueB = [&](int kk, int buf) {
        // 64 rows x 64B tile = 256 x b128 transfers, 2 per thread
#pragma unroll
        for (int t = 0; t < 2; ++t) {
            const int id  = tid + t * 128;     // 0..255
            const int row = id >> 2;
            const int seg = id & 3;
            const _Float16* g = W + (long)(n0base + row) * ldw + kk * 32 + seg * 8;
            _Float16* l = &bsm[buf][row * 32 + seg * 8];
            __builtin_amdgcn_global_load_async_to_lds_b128(
                (as1_v4i*)g, (as3_v4i*)l, 0, 0);
        }
    };
    issueB(0, 0);
    __builtin_amdgcn_s_wait_asynccnt(0);
    __syncthreads();
    for (int kk = 0; kk < ksteps; ++kk) {
        const int buf = kk & 1;
        if (kk + 1 < ksteps) issueB(kk + 1, buf ^ 1);   // prefetch next tile
        v16h a = load_a_frag(A, lda, rowA, kk * 32, lane);
#pragma unroll
        for (int t = 0; t < 4; ++t) {
            const _Float16* br = &bsm[buf][(t * 16 + (lane & 15)) * 32 + ((lane >> 4) * 16)];
            v16h b = *(const v16h*)br;                  // ds_load
            acc[t] = __builtin_amdgcn_wmma_f32_16x16x32_f16(
                false, a, false, b, (short)0, acc[t], false, false);
        }
        __builtin_amdgcn_s_wait_asynccnt(0);            // prefetched tile landed
        __syncthreads();                                // safe to reuse buffers
    }
#else
    for (int kk = 0; kk < ksteps; ++kk) {
        v16h a = load_a_frag(A, lda, rowA, kk * 32, lane);
#pragma unroll
        for (int t = 0; t < 4; ++t) {
            v16h b = load_b_frag(W, ldw, n0base + 16 * t, kk * 32, lane);
            acc[t] = __builtin_amdgcn_wmma_f32_16x16x32_f16(
                false, a, false, b, (short)0, acc[t], false, false);
        }
    }
#endif

    if (active) {
        // D 16x16 f32: VGPR v, lanes 0-15 -> (M=v,N=lane); 16-31 -> (M=8+v,N=lane-16)
        const int mbase = row0 + (lane >> 4) * 8;
        const int col   = (lane & 15);
#pragma unroll
        for (int t = 0; t < 4; ++t) {
            const int c0 = n0base + 16 * t + col;
#pragma unroll
            for (int v = 0; v < 8; ++v) {
                float val = acc[t][v];
                if (EPI == 1) {
                    val += bias[(long)dir * strideBias + c0];
                    val = (val > 20.f) ? val : __logf(1.f + __expf(val));  // softplus
                }
                C[(long)(mbase + v) * ldc + c0] = val;
            }
        }
    }
}

// ---------------------------------------------------------------------------
__global__ void f32_to_f16_kernel(const float* __restrict__ in,
                                  _Float16* __restrict__ out, long n) {
    long i = (long)blockIdx.x * blockDim.x + threadIdx.x;
    if (i < n) out[i] = (_Float16)in[i];
}

// Scatter x (B, DM, D,H,W) into 6 permuted sequences xseq[dir][b][l][c] (f16)
__global__ void build_xseq_kernel(const float* __restrict__ x,
                                  _Float16* __restrict__ xseq) {
    long idx = (long)blockIdx.x * blockDim.x + threadIdx.x;
    const long total = (long)NDIR * NB * LSEQ * DM;
    if (idx >= total) return;
    int c   = (int)(idx & (DM - 1));
    long r  = idx >> 9;
    int p   = (int)(r % LSEQ);
    long db = r / LSEQ;
    int b   = (int)(db % NB);
    int dir = (int)(db / NB);
    int d = p / 100, h = (p / 10) % 10, w = p % 10;
    int l;
    switch (dir) {
        case 0: l = p; break;
        case 1: l = d * 100 + w * 10 + h; break;
        case 2: l = (9 - w) * 100 + h * 10 + d; break;
        case 3: l = 999 - p; break;
        case 4: l = 999 - (d * 100 + w * 10 + h); break;
        default: l = 999 - ((9 - w) * 100 + h * 10 + d); break;
    }
    float v = x[((long)b * DM + c) * LSEQ + p];
    xseq[(((long)dir * NB + b) * LSEQ + l) * DM + c] = (_Float16)v;
}

// Depthwise causal conv(4) + bias + SiLU on xz[..., :DI]; writes xt (f16)
__global__ void conv_silu_kernel(const float* __restrict__ xz,
                                 const float* __restrict__ cw,
                                 const float* __restrict__ cb,
                                 _Float16* __restrict__ xt) {
    long idx = (long)blockIdx.x * blockDim.x + threadIdx.x;
    const long total = (long)NDIR * BL * DI;
    if (idx >= total) return;
    int d   = (int)(idx & (DI - 1));
    long bl = idx >> 10;
    int l   = (int)(bl % LSEQ);
    long db = bl / LSEQ;
    int dir = (int)(db / NB);
    const float* w = cw + ((long)dir * DI + d) * 4;
    float acc = cb[(long)dir * DI + d];
    const float* base = xz + (bl - l) * (2 * DI) + d;
#pragma unroll
    for (int k = 0; k < 4; ++k) {
        int ls = l - 3 + k;
        if (ls >= 0) acc += w[k] * base[(long)ls * (2 * DI)];
    }
    float s = acc * (1.f / (1.f + __expf(-acc)));      // SiLU
    xt[idx] = (_Float16)s;
}

// dt-proj input slice: first DTR columns of xdbl, as f16
__global__ void slice_dtin_kernel(const float* __restrict__ xdbl,
                                  _Float16* __restrict__ dtin) {
    long idx = (long)blockIdx.x * blockDim.x + threadIdx.x;
    const long total = (long)NDIR * BL * DTR;
    if (idx >= total) return;
    int c = (int)(idx & (DTR - 1));
    long row = idx >> 5;
    dtin[idx] = (_Float16)xdbl[row * (DTR + 2 * DS) + c];
}

// -------------------- chunked selective scan (3 passes) --------------------
// Pass 1: per (dir,b,chunk,d) run the recurrence from h=0 over LC steps,
// recording final local state h and cumulative decay prod(dA) per state.
__global__ void scan_pass1_kernel(const _Float16* __restrict__ xt,
                                  const float* __restrict__ dt,
                                  const float* __restrict__ xdbl,
                                  const float* __restrict__ A_log,
                                  float* __restrict__ chk_h,
                                  float* __restrict__ chk_pA) {
    int idx = blockIdx.x * blockDim.x + threadIdx.x;   // NDIR*NB*CH*DI = 98304
    int d   = idx & (DI - 1);
    int c   = (idx >> 10) & (CH - 1);
    int db  = idx >> 13;
    int dir = db / NB;
    float As[DS];
#pragma unroll
    for (int s = 0; s < DS; ++s)
        As[s] = -__expf(A_log[((long)dir * DI + d) * DS + s]);
    float h[DS], pA[DS];
#pragma unroll
    for (int s = 0; s < DS; ++s) { h[s] = 0.f; pA[s] = 1.f; }

    const long rowbase = (long)db * LSEQ + c * LC;
    for (int l = 0; l < LC; ++l) {
        const long row = rowbase + l;
        const float xv  = (float)xt[row * DI + d];
        const float dtv = dt[row * DI + d];
        const float* xr = xdbl + row * (DTR + 2 * DS);
        const float dtx = dtv * xv;
#pragma unroll
        for (int s = 0; s < DS; ++s) {
            float dA = __expf(dtv * As[s]);
            h[s]  = dA * h[s] + dtx * xr[DTR + s];
            pA[s] *= dA;
        }
    }
    const long base = ((long)(db * CH + c) * DI + d) * DS;
#pragma unroll
    for (int s = 0; s < DS; ++s) { chk_h[base + s] = h[s]; chk_pA[base + s] = pA[s]; }
}

// Pass 2: sequential 8-step carry combine per (dir,b,d); writes the entry
// state for every chunk.  carry[c+1] = pA[c] * carry[c] + h[c]
__global__ void scan_combine_kernel(const float* __restrict__ chk_h,
                                    const float* __restrict__ chk_pA,
                                    float* __restrict__ carry) {
    int idx = blockIdx.x * blockDim.x + threadIdx.x;   // NDIR*NB*DI = 12288
    int d  = idx & (DI - 1);
    int db = idx >> 10;
    float cr[DS];
#pragma unroll
    for (int s = 0; s < DS; ++s) cr[s] = 0.f;
    for (int c = 0; c < CH; ++c) {
        const long base = ((long)(db * CH + c) * DI + d) * DS;
#pragma unroll
        for (int s = 0; s < DS; ++s) carry[base + s] = cr[s];
#pragma unroll
        for (int s = 0; s < DS; ++s)
            cr[s] = chk_pA[base + s] * cr[s] + chk_h[base + s];
    }
}

// Pass 3: re-run each chunk from its carry, producing y, skip, SiLU(z) gate.
__global__ void scan_pass3_kernel(const _Float16* __restrict__ xt,
                                  const float* __restrict__ dt,
                                  const float* __restrict__ xdbl,
                                  const float* __restrict__ xz,
                                  const float* __restrict__ A_log,
                                  const float* __restrict__ D_param,
                                  const float* __restrict__ carry,
                                  _Float16* __restrict__ ygate) {
    int idx = blockIdx.x * blockDim.x + threadIdx.x;   // 98304
    int d   = idx & (DI - 1);
    int c   = (idx >> 10) & (CH - 1);
    int db  = idx >> 13;
    int dir = db / NB;
    float As[DS];
#pragma unroll
    for (int s = 0; s < DS; ++s)
        As[s] = -__expf(A_log[((long)dir * DI + d) * DS + s]);
    const float Dp = D_param[(long)dir * DI + d];
    float h[DS];
    const long cbase = ((long)(db * CH + c) * DI + d) * DS;
#pragma unroll
    for (int s = 0; s < DS; ++s) h[s] = carry[cbase + s];

    const long rowbase = (long)db * LSEQ + c * LC;
    for (int l = 0; l < LC; ++l) {
        const long row = rowbase + l;
        const float xv  = (float)xt[row * DI + d];
        const float dtv = dt[row * DI + d];
        const float* xr = xdbl + row * (DTR + 2 * DS);
        const float dtx = dtv * xv;
        float y = 0.f;
#pragma unroll
        for (int s = 0; s < DS; ++s) {
            float dA = __expf(dtv * As[s]);
            h[s] = dA * h[s] + dtx * xr[DTR + s];        // B_c
            y   += h[s] * xr[DTR + DS + s];              // C_c
        }
        y += Dp * xv;
        const float zv = xz[row * (2 * DI) + DI + d];
        y *= zv * (1.f / (1.f + __expf(-zv)));           // * SiLU(z)
        ygate[row * DI + d] = (_Float16)y;
    }
}

// Gather-sum with inverse permutations: out[b][c][p] = mean_dir ydir[dir][b, l_dir(p), c]
__global__ void final_sum_kernel(const float* __restrict__ ydir,
                                 float* __restrict__ out) {
    long idx = (long)blockIdx.x * blockDim.x + threadIdx.x;
    const long total = (long)NB * DM * LSEQ;
    if (idx >= total) return;
    int p   = (int)(idx % LSEQ);
    long bc = idx / LSEQ;
    int c   = (int)(bc & (DM - 1));
    int b   = (int)(bc >> 9);
    int d = p / 100, h = (p / 10) % 10, w = p % 10;
    int l[NDIR];
    l[0] = p;
    l[1] = d * 100 + w * 10 + h;
    l[2] = (9 - w) * 100 + h * 10 + d;
    l[3] = 999 - l[0];
    l[4] = 999 - l[1];
    l[5] = 999 - l[2];
    float s = 0.f;
#pragma unroll
    for (int dir = 0; dir < NDIR; ++dir)
        s += ydir[(((long)dir * NB + b) * LSEQ + l[dir]) * DM + c];
    out[idx] = s * (1.f / 6.f);
}

// ---------------------------------------------------------------------------
extern "C" void kernel_launch(void* const* d_in, const int* in_sizes, int n_in,
                              void* d_out, int out_size, void* d_ws, size_t ws_size,
                              hipStream_t stream) {
    (void)in_sizes; (void)n_in; (void)out_size; (void)ws_size;
    const float* x    = (const float*)d_in[0];
    const float* ipw  = (const float*)d_in[1];   // (6, 2048, 512)
    const float* cw   = (const float*)d_in[2];   // (6, 1024, 1, 4)
    const float* cb   = (const float*)d_in[3];   // (6, 1024)
    const float* xpw  = (const float*)d_in[4];   // (6, 64, 1024)
    const float* dpw  = (const float*)d_in[5];   // (6, 1024, 32)
    const float* dpb  = (const float*)d_in[6];   // (6, 1024)
    const float* alog = (const float*)d_in[7];   // (6, 1024, 16)
    const float* dpar = (const float*)d_in[8];   // (6, 1024)
    const float* opw  = (const float*)d_in[9];   // (6, 512, 1024)
    float* out = (float*)d_out;

    size_t off = 0;
    auto carve = [&](size_t bytes) -> void* {
        void* p = (char*)d_ws + off;
        off = (off + bytes + 255) & ~(size_t)255;
        return p;
    };
    const long nIPW = (long)NDIR * 2 * DI * DM;
    const long nXPW = (long)NDIR * (DTR + 2 * DS) * DI;
    const long nDPW = (long)NDIR * DI * DTR;
    const long nOPW = (long)NDIR * DM * DI;
    const long nCHK = (long)NDIR * NB * CH * DI * DS;
    _Float16* w_ip   = (_Float16*)carve(nIPW * 2);
    _Float16* w_xp   = (_Float16*)carve(nXPW * 2);
    _Float16* w_dp   = (_Float16*)carve(nDPW * 2);
    _Float16* w_op   = (_Float16*)carve(nOPW * 2);
    _Float16* xseq   = (_Float16*)carve((long)NDIR * BL * DM * 2);
    float*    xz     = (float*)   carve((long)NDIR * BL * 2 * DI * 4);
    _Float16* xt     = (_Float16*)carve((long)NDIR * BL * DI * 2);
    float*    xdbl   = (float*)   carve((long)NDIR * BL * (DTR + 2 * DS) * 4);
    _Float16* dtin   = (_Float16*)carve((long)NDIR * BL * DTR * 2);
    float*    dtbuf  = (float*)   carve((long)NDIR * BL * DI * 4);
    _Float16* ygate  = (_Float16*)carve((long)NDIR * BL * DI * 2);
    float*    ydir   = (float*)   carve((long)NDIR * BL * DM * 4);
    float*    chk_h  = (float*)   carve(nCHK * 4);
    float*    chk_pA = (float*)   carve(nCHK * 4);
    float*    cbuf   = (float*)   carve(nCHK * 4);

    auto cdiv = [](long a, long b) { return (int)((a + b - 1) / b); };
    const int gx = cdiv(BL / 16, 4);   // 4 M-tiles (waves) per block

    f32_to_f16_kernel<<<cdiv(nIPW, 256), 256, 0, stream>>>(ipw, w_ip, nIPW);
    f32_to_f16_kernel<<<cdiv(nXPW, 256), 256, 0, stream>>>(xpw, w_xp, nXPW);
    f32_to_f16_kernel<<<cdiv(nDPW, 256), 256, 0, stream>>>(dpw, w_dp, nDPW);
    f32_to_f16_kernel<<<cdiv(nOPW, 256), 256, 0, stream>>>(opw, w_op, nOPW);

    build_xseq_kernel<<<cdiv((long)NDIR * BL * DM, 256), 256, 0, stream>>>(x, xseq);

    // GEMM1: xz = xseq (2000x512) * ipw^T (512x2048)
    gemm_wmma_f16<0><<<dim3(gx, (2 * DI) / 64, NDIR), 128, 0, stream>>>(
        xseq, (long)BL * DM, DM, w_ip, (long)2 * DI * DM, DM,
        xz, (long)BL * 2 * DI, 2 * DI, nullptr, 0, BL, 2 * DI, DM);

    conv_silu_kernel<<<cdiv((long)NDIR * BL * DI, 256), 256, 0, stream>>>(xz, cw, cb, xt);

    // GEMM2: xdbl = xt (2000x1024) * xpw^T (1024x64)
    gemm_wmma_f16<0><<<dim3(gx, (DTR + 2 * DS) / 64, NDIR), 128, 0, stream>>>(
        xt, (long)BL * DI, DI, w_xp, (long)(DTR + 2 * DS) * DI, DI,
        xdbl, (long)BL * (DTR + 2 * DS), DTR + 2 * DS, nullptr, 0,
        BL, DTR + 2 * DS, DI);

    slice_dtin_kernel<<<cdiv((long)NDIR * BL * DTR, 256), 256, 0, stream>>>(xdbl, dtin);

    // GEMM3 (+bias+softplus): dt = softplus(dtin (2000x32) * dpw^T (32x1024) + dpb)
    gemm_wmma_f16<1><<<dim3(gx, DI / 64, NDIR), 128, 0, stream>>>(
        dtin, (long)BL * DTR, DTR, w_dp, (long)DI * DTR, DTR,
        dtbuf, (long)BL * DI, DI, dpb, DI, BL, DI, DTR);

    // chunked scan: local pass, carry combine, final pass (+skip+gate)
    scan_pass1_kernel<<<(NDIR * NB * CH * DI) / 256, 256, 0, stream>>>(
        xt, dtbuf, xdbl, alog, chk_h, chk_pA);
    scan_combine_kernel<<<(NDIR * NB * DI) / 256, 256, 0, stream>>>(
        chk_h, chk_pA, cbuf);
    scan_pass3_kernel<<<(NDIR * NB * CH * DI) / 256, 256, 0, stream>>>(
        xt, dtbuf, xdbl, xz, alog, dpar, cbuf, ygate);

    // GEMM4: ydir = ygate (2000x1024) * opw^T (1024x512)
    gemm_wmma_f16<0><<<dim3(gx, DM / 64, NDIR), 128, 0, stream>>>(
        ygate, (long)BL * DI, DI, w_op, (long)DM * DI, DI,
        ydir, (long)BL * DM, DM, nullptr, 0, BL, DM, DI);

    final_sum_kernel<<<cdiv((long)NB * DM * LSEQ, 256), 256, 0, stream>>>(ydir, out);
}
